// LSTM_32985348833681
// MI455X (gfx1250) — matrix operand order repaired and checked by reading, not compiled, not run on hardware
//
#include <hip/hip_runtime.h>
#include <hip/hip_bf16.h>
#include <math.h>

// ---------------- problem constants ----------------
constexpr int Bb   = 64;      // batch
constexpr int Tt   = 2048;    // timesteps
constexpr int DIN  = 256;
constexpr int DH   = 256;
constexpr int DC   = 256;
constexpr int DOUT = 256;
constexpr int DK   = DIN + DH;        // 512 (K for gate GEMMs)

constexpr int NWG  = 16;              // persistent workgroups
constexpr int NC   = DC / NWG;        // 16 columns of each gate / c / h / out per WG
constexpr int NGC  = 3 * NC;          // 48 gate columns per WG (f|i|o)

// tiles
constexpr int MT   = Bb / 16;         // 4 M-tiles
constexpr int KT_G = DK / 32;         // 16 K-steps (gates)
constexpr int KT_O = DH / 32;         // 8  K-steps (out GEMM)

// ---------------- LDS layout (bytes) ----------------
constexpr int L_COMB  = 0;                               // 64 x 512 bf16  (x_t | h_t), row-major
constexpr int L_W     = L_COMB + Bb * DK * 2;            // 48 x 512 bf16  weight slice, K-contiguous
constexpr int L_WOUT  = L_W + NGC * DK * 2;              // 16 x 256 bf16  Wout slice, K-contiguous
constexpr int L_GATES = L_WOUT + NC * DH * 2;            // 64 x 48 f32    gate pre-activations
constexpr int L_C     = L_GATES + Bb * NGC * 4;          // 64 x 16 f32    cell slice
constexpr int L_BIAS  = L_C + Bb * NC * 4;               // 64 f32         bf|bi|bo|bout slices
constexpr int SMEM_BYTES = L_BIAS + 64 * 4;              // ~139.5 KB

// ---------------- workspace layout (bytes) ----------------
constexpr size_t WS_WCAT = 0;                                  // 768 x 512 bf16 (rows = gate*256+col, K-contig)
constexpr size_t WS_WOUT = WS_WCAT + (size_t)3 * DC * DK * 2;  // 256 x 256 bf16 (rows = col, K-contig)
constexpr size_t WS_HBUF = WS_WOUT + (size_t)DOUT * DH * 2;    // 2 x 64 x 256 bf16 ping-pong h
constexpr size_t WS_CNT  = WS_HBUF + (size_t)2 * Bb * DH * 2;  // (T+2) u32 barrier counters

// output offsets (elements)
constexpr size_t OUT_ELEMS = (size_t)Bb * Tt * DOUT;
constexpr size_t HOFF = OUT_ELEMS;
constexpr size_t COFF = HOFF + (size_t)Bb * DH;

typedef __attribute__((ext_vector_type(16))) __bf16 bf16x16;
typedef __attribute__((ext_vector_type(8)))  __bf16 bf16x8;
typedef __attribute__((ext_vector_type(8)))  float  f32x8;
typedef __attribute__((ext_vector_type(4)))  int    i32x4;
typedef __attribute__((address_space(1)))    i32x4  gi32x4;   // global int4
typedef __attribute__((address_space(3)))    i32x4  li32x4;   // LDS int4

#if defined(__HIP_DEVICE_COMPILE__) && defined(__has_builtin)
#if __has_builtin(__builtin_amdgcn_global_load_async_to_lds_b128) && \
    __has_builtin(__builtin_amdgcn_s_wait_asynccnt)
#define USE_ASYNC_LDS 1
#endif
#endif

__device__ __forceinline__ float sigmoidf_(float x) { return 1.0f / (1.0f + __expf(-x)); }

// A fragment (16x32 bf16): base is row-major with row stride 512 halves.
// lane<16: row=lane,   K = kb+0..7  and kb+16..23
// lane>=16: row=lane-16,K = kb+8..15 and kb+24..31
__device__ __forceinline__ bf16x16 load_a_frag(const __bf16* base, int mt, int kt) {
  int lane = threadIdx.x & 31;
  int row  = mt * 16 + (lane & 15);
  int kb   = kt * 32 + ((lane >> 4) << 3);
  const __bf16* p = base + row * DK + kb;
  bf16x8 lo = *(const bf16x8*)(p);
  bf16x8 hi = *(const bf16x8*)(p + 16);
  bf16x16 a;
#pragma unroll
  for (int i = 0; i < 8; ++i) { a[i] = lo[i]; a[i + 8] = hi[i]; }
  return a;
}

// B fragment (32x16 bf16): base is column-major (K-contiguous) with column stride `kstride` halves.
// lane<16: col=lane,    K = kb+0..7 and kb+16..23 ; lane>=16: col=lane-16, K = kb+8..15 and kb+24..31
__device__ __forceinline__ bf16x16 load_b_frag(const __bf16* base, int nt, int kt, int kstride) {
  int lane = threadIdx.x & 31;
  int col  = nt * 16 + (lane & 15);
  int kb   = kt * 32 + ((lane >> 4) << 3);
  const __bf16* p = base + col * kstride + kb;
  bf16x8 lo = *(const bf16x8*)(p);
  bf16x8 hi = *(const bf16x8*)(p + 16);
  bf16x16 b;
#pragma unroll
  for (int i = 0; i < 8; ++i) { b[i] = lo[i]; b[i + 8] = hi[i]; }
  return b;
}

// ---------------- pre-pass: fp32 -> bf16 weight transpose + h0 init ----------------
__global__ void lstm_prep_kernel(const float* __restrict__ Wf, const float* __restrict__ Wi,
                                 const float* __restrict__ Wo, const float* __restrict__ Wout,
                                 const float* __restrict__ h0,
                                 __bf16* __restrict__ WcatT, __bf16* __restrict__ WoutT,
                                 __bf16* __restrict__ hbuf0) {
  int idx = blockIdx.x * 256 + threadIdx.x;
  if (idx < 3 * DC * DK) {                        // WcatT[(gate*256+col)*512 + k] = Wg[k][col]
    int row  = idx >> 9;                          // gate*256+col
    int k    = idx & (DK - 1);
    int gate = row >> 8;
    int col  = row & 255;
    const float* W = (gate == 0) ? Wf : ((gate == 1) ? Wi : Wo);
    WcatT[idx] = (__bf16)W[k * DC + col];
  }
  int i2 = idx - 3 * DC * DK;
  if (i2 >= 0 && i2 < DOUT * DH) {                // WoutT[col*256 + k] = Wout[k][col]
    int col = i2 >> 8;
    int k   = i2 & 255;
    WoutT[i2] = (__bf16)Wout[k * DOUT + col];
  }
  int i3 = idx - 3 * DC * DK - DOUT * DH;
  if (i3 >= 0 && i3 < Bb * DH) hbuf0[i3] = (__bf16)h0[i3];
}

// ---------------- inter-workgroup release/acquire barrier ----------------
__device__ __forceinline__ void grid_barrier(unsigned* cnt) {
  __syncthreads();
  if (threadIdx.x == 0) {
    __threadfence();                       // release: make hbuf stores visible
    atomicAdd(cnt, 1u);
    volatile unsigned* vc = cnt;
    while (*vc < (unsigned)NWG) { __builtin_amdgcn_s_sleep(2); }
  }
  __syncthreads();
  __threadfence();                         // acquire
}

// ---------------- persistent LSTM kernel: 16 WGs, each owns 16 columns ----------------
__global__ void __launch_bounds__(256, 1)
lstm_persistent_kernel(const float* __restrict__ x, const float* __restrict__ cell,
                       const float* __restrict__ bf, const float* __restrict__ bi,
                       const float* __restrict__ bo, const float* __restrict__ bout,
                       const __bf16* __restrict__ WcatT, const __bf16* __restrict__ WoutT,
                       __bf16* __restrict__ hbuf, unsigned* __restrict__ cnt,
                       float* __restrict__ out) {
  extern __shared__ char smem[];
  __bf16* comb   = (__bf16*)(smem + L_COMB);    // [64][512]
  __bf16* wlds   = (__bf16*)(smem + L_W);       // [48][512] K-contig
  __bf16* wouts  = (__bf16*)(smem + L_WOUT);    // [16][256] K-contig
  float*  gates  = (float*) (smem + L_GATES);   // [64][48]
  float*  cbuf   = (float*) (smem + L_C);       // [64][16]
  float*  bias   = (float*) (smem + L_BIAS);    // [64]  f|i|o|out

  const int tid = threadIdx.x;
  const int wg  = blockIdx.x;                   // 0..15
  const int w   = tid >> 5;                     // wave id 0..7

  // ---- one-time init: weight slices -> LDS (vectorized), cell slice, biases ----
  {
    // 3 gate chunks, each 16 contiguous rows x 512 halves = 1024 uint4
    for (int i = tid; i < 3 * 1024; i += 256) {
      int g = i >> 10, r = i & 1023;
      const uint4* src = (const uint4*)(WcatT + (size_t)(g * DC + wg * NC) * DK);
      *(uint4*)(wlds + g * NC * DK + r * 8) = src[r];
    }
    // Wout slice: 16 rows x 256 halves = 512 uint4
    for (int i = tid; i < 512; i += 256) {
      const uint4* src = (const uint4*)(WoutT + (size_t)(wg * NC) * DH);
      *(uint4*)(wouts + i * 8) = src[i];
    }
    for (int i = tid; i < Bb * NC; i += 256) {
      int row = i >> 4, col = i & 15;
      cbuf[i] = cell[row * DC + wg * NC + col];
    }
    if (tid < NC) {
      bias[tid]      = bf[wg * NC + tid];
      bias[16 + tid] = bi[wg * NC + tid];
      bias[32 + tid] = bo[wg * NC + tid];
      bias[48 + tid] = bout[wg * NC + tid];
    }
  }

  // ---- time loop (t == Tt is an epilogue that only computes out[T-1]) ----
  for (int t = 0; t <= Tt; ++t) {
    grid_barrier(cnt + t);                                  // h_t published by all WGs

    const __bf16* hb_cur  = hbuf + (size_t)(t & 1) * (Bb * DH);
    __bf16*       hb_next = hbuf + (size_t)((t + 1) & 1) * (Bb * DH);

    // ---- load comb = [x_t | h_t] into LDS ----
    if (t < Tt) {
      for (int i = tid; i < Bb * (DIN / 4); i += 256) {     // x part: 64 rows x 64 float4
        int b = i >> 6, j4 = i & 63;
        const float* xp = x + (size_t)b * Tt * DIN + (size_t)t * DIN + j4 * 4;
        float4 v = *(const float4*)xp;
        if (t + 1 < Tt) __builtin_prefetch(xp + DIN, 0, 1); // next step's x -> cache
        __bf16* d = comb + b * DK + j4 * 4;
        d[0] = (__bf16)v.x; d[1] = (__bf16)v.y; d[2] = (__bf16)v.z; d[3] = (__bf16)v.w;
      }
    }
#if defined(USE_ASYNC_LDS)
    // h part: 2048 x 16B direct global->LDS async copies (ASYNCcnt-tracked, no VGPR data)
    for (int i = tid; i < (Bb * DH) / 8; i += 256) {
      int b = i >> 5, j = i & 31;
      __builtin_amdgcn_global_load_async_to_lds_b128(
          (gi32x4*)(hb_cur + (size_t)i * 8),
          (li32x4*)(comb + b * DK + DIN + j * 8),
          0, 0);
    }
    __builtin_amdgcn_s_wait_asynccnt(0);
#else
    for (int i = tid; i < (Bb * DH) / 8; i += 256) {        // h part: 2048 uint4
      int b = i >> 5, j = i & 31;
      *(uint4*)(comb + b * DK + DIN + j * 8) = ((const uint4*)hb_cur)[i];
    }
#endif
    __syncthreads();

    // ---- wave-specialized GEMMs ----
    if (w < MT) {
      // waves 0..3: gate N-tiles {0,2} for mt=w, sharing one A-fragment per K-step
      if (t < Tt) {
        const int mt = w;
        f32x8 acc0 = {}, acc2 = {};
#pragma unroll
        for (int kt = 0; kt < KT_G; ++kt) {
          bf16x16 a  = load_a_frag(comb, mt, kt);
          bf16x16 b0 = load_b_frag(wlds, 0, kt, DK);
          acc0 = __builtin_amdgcn_wmma_f32_16x16x32_bf16(false, a, false, b0,
                                                         (short)0, acc0, false, false);
          bf16x16 b2 = load_b_frag(wlds, 2, kt, DK);
          acc2 = __builtin_amdgcn_wmma_f32_16x16x32_bf16(false, a, false, b2,
                                                         (short)0, acc2, false, false);
        }
        int lane = tid & 31;
        int colB = lane & 15;
        int rowB = mt * 16 + ((lane >> 4) << 3);
#pragma unroll
        for (int r = 0; r < 8; ++r) {
          gates[(rowB + r) * NGC + colB]           = acc0[r];
          gates[(rowB + r) * NGC + 2 * NC + colB]  = acc2[r];
        }
      }
    } else {
      const int mt = w - MT;                                // 0..3
      // out_{t-1} = sigmoid(h_t @ Wout_slice + bout_slice)
      if (t > 0) {
        f32x8 acc = {};
#pragma unroll
        for (int kt = 0; kt < KT_O; ++kt) {
          bf16x16 a = load_a_frag(comb + DIN, mt, kt);      // h part of comb
          bf16x16 b = load_b_frag(wouts, 0, kt, DH);
          acc = __builtin_amdgcn_wmma_f32_16x16x32_bf16(false, a, false, b,
                                                        (short)0, acc, false, false);
        }
        int lane = tid & 31;
        int col  = lane & 15;
        int rowB = mt * 16 + ((lane >> 4) << 3);
        float bo_ = bias[48 + col];
#pragma unroll
        for (int r = 0; r < 8; ++r) {
          float v = sigmoidf_(acc[r] + bo_);
          out[(size_t)(rowB + r) * Tt * DOUT + (size_t)(t - 1) * DOUT + wg * NC + col] = v;
        }
      }
      // gate N-tile {1} for mt
      if (t < Tt) {
        f32x8 acc1 = {};
#pragma unroll
        for (int kt = 0; kt < KT_G; ++kt) {
          bf16x16 a  = load_a_frag(comb, mt, kt);
          bf16x16 b1 = load_b_frag(wlds, 1, kt, DK);
          acc1 = __builtin_amdgcn_wmma_f32_16x16x32_bf16(false, a, false, b1,
                                                         (short)0, acc1, false, false);
        }
        int lane = tid & 31;
        int colB = NC + (lane & 15);
        int rowB = mt * 16 + ((lane >> 4) << 3);
#pragma unroll
        for (int r = 0; r < 8; ++r) gates[(rowB + r) * NGC + colB] = acc1[r];
      }
    }

    if (t < Tt) {
      __syncthreads();

      // ---- elementwise cell/h update for this WG's 16 columns (64x16 = 4 per thread)
      for (int e = tid; e < Bb * NC; e += 256) {
        int row = e >> 4, col = e & 15;
        float fpre = gates[row * NGC + col]           + bias[col];
        float ipre = gates[row * NGC + NC + col]      + bias[16 + col];
        float opre = gates[row * NGC + 2 * NC + col]  + bias[32 + col];
        float fg = sigmoidf_(fpre);
        float ig = sigmoidf_(ipre);
        float z  = tanhf(ipre);
        float og = sigmoidf_(opre);
        float c  = cbuf[e];
        float cn = c * fg + z * ig;
        float hn = tanhf(cn) * og;
        cbuf[e] = cn;
        hb_next[row * DH + wg * NC + col] = (__bf16)hn;
        if (t == Tt - 1) {
          out[HOFF + (size_t)row * DH + wg * NC + col] = hn;   // h_final (fp32 precision)
          out[COFF + (size_t)row * DC + wg * NC + col] = cn;   // c_final
        }
      }
    }
  }
}

extern "C" void kernel_launch(void* const* d_in, const int* in_sizes, int n_in,
                              void* d_out, int out_size, void* d_ws, size_t ws_size,
                              hipStream_t stream) {
  const float* x    = (const float*)d_in[0];
  const float* h0   = (const float*)d_in[1];
  const float* cell = (const float*)d_in[2];
  const float* Wf   = (const float*)d_in[3];
  const float* bf   = (const float*)d_in[4];
  const float* Wi   = (const float*)d_in[5];
  const float* bi   = (const float*)d_in[6];
  const float* Wo   = (const float*)d_in[7];
  const float* bo   = (const float*)d_in[8];
  const float* Wout = (const float*)d_in[9];
  const float* bout = (const float*)d_in[10];
  float* out = (float*)d_out;

  char* ws = (char*)d_ws;
  __bf16*   WcatT = (__bf16*)(ws + WS_WCAT);
  __bf16*   WoutT = (__bf16*)(ws + WS_WOUT);
  __bf16*   hbuf  = (__bf16*)(ws + WS_HBUF);
  unsigned* cnt   = (unsigned*)(ws + WS_CNT);

  // zero per-step barrier counters every call (graph-capture safe)
  (void)hipMemsetAsync(cnt, 0, (size_t)(Tt + 2) * sizeof(unsigned), stream);

  // weight transpose/convert + h0 init
  int prep_work = 3 * DC * DK + DOUT * DH + Bb * DH;
  lstm_prep_kernel<<<(prep_work + 255) / 256, 256, 0, stream>>>(
      Wf, Wi, Wo, Wout, h0, WcatT, WoutT, hbuf);

  // persistent LSTM: 16 co-resident WGs, ~136KB dynamic LDS each
  (void)hipFuncSetAttribute((const void*)lstm_persistent_kernel,
                            hipFuncAttributeMaxDynamicSharedMemorySize, SMEM_BYTES);
  lstm_persistent_kernel<<<dim3(NWG), dim3(256), SMEM_BYTES, stream>>>(
      x, cell, bf, bi, bo, bout, WcatT, WoutT, hbuf, cnt, out);
}
